// Node_Edge_Switch_Attention_26259430048706
// MI455X (gfx1250) — matrix operand order repaired and checked by reading, not compiled, not run on hardware
//
#include <hip/hip_runtime.h>
#include <hip/hip_bf16.h>

typedef __attribute__((ext_vector_type(16))) _Float16 v16h;
typedef __attribute__((ext_vector_type(8)))  _Float16 v8h;
typedef __attribute__((ext_vector_type(4)))  _Float16 v4h;
typedef __attribute__((ext_vector_type(8)))  float    v8f;

#define N_HEADS 8
#define HIDDEN  256
#define D_HEAD  32
#define N_NODES 4096
#define N_EDGES 16384
#define ATT_WAVES 4
#define KSPLIT  4
#define KCHUNK  (N_EDGES / KSPLIT)   // 4096

// ---- WMMA 16x16x32 f16 lane/element maps (CDNA5 ISA 7.12.2) ----
// A (16x32): lane%16 = M row; per lane two contiguous K-runs of 8: {hi*8..} and {16+hi*8..}
// B (32x16): lane%16 = N col; per lane same K-runs as A: {hi*8..} and {16+hi*8..}
// C/D (16x16 f32): element r -> row = r + hi*8, col = lane%16

__device__ __forceinline__ float redmax16(float v) {
    v = fmaxf(v, __shfl_xor(v, 1, 16));
    v = fmaxf(v, __shfl_xor(v, 2, 16));
    v = fmaxf(v, __shfl_xor(v, 4, 16));
    v = fmaxf(v, __shfl_xor(v, 8, 16));
    return v;
}
__device__ __forceinline__ float redsum16(float v) {
    v += __shfl_xor(v, 1, 16);
    v += __shfl_xor(v, 2, 16);
    v += __shfl_xor(v, 4, 16);
    v += __shfl_xor(v, 8, 16);
    return v;
}

// ---------------- row sums of T -> 1/rowsum (1.0 if row sums to 0) ----------------
__global__ void rowsum_inv_kernel(const float* __restrict__ T, float* __restrict__ inv) {
    const int n = blockIdx.x;
    const int tid = threadIdx.x; // 256 threads
    const float4* row = (const float4*)(T + (size_t)n * N_EDGES);
    float s = 0.f;
    for (int i = tid; i < N_EDGES / 4; i += 256) {
        float4 t = row[i];
        s += t.x + t.y + t.z + t.w;
    }
    __shared__ float red[256];
    red[tid] = s;
    __syncthreads();
    for (int w = 128; w > 0; w >>= 1) {
        if (tid < w) red[tid] += red[tid + w];
        __syncthreads();
    }
    if (tid == 0) {
        float rs = red[0];
        inv[n] = (rs == 0.f) ? 1.f : (1.f / rs);
    }
}

// ---------------- Y[M,256] = X[M,256] @ W^T + bias, K=256 ----------------
// TROUT: write Y transposed ([256][M], f16) for contiguous B-fragment loads downstream.
template <typename XT, typename YT, bool TROUT>
__global__ void gemm256_wmma_kernel(const XT* __restrict__ X, const float* __restrict__ W,
                                    const float* __restrict__ bias, YT* __restrict__ Y, int M) {
    const int lane = threadIdx.x & 31;
    const int wid  = (blockIdx.x * blockDim.x + threadIdx.x) >> 5;
    const int mt = wid / 16;
    const int nt = wid % 16;
    const int hi  = (lane & 16) ? 1 : 0;
    const int l15 = lane & 15;

    const int arow = mt * 16 + l15;
    const int bn   = nt * 16 + l15;   // row of W (Y = X @ W^T)

    v8f c = {};
#pragma unroll
    for (int kk = 0; kk < HIDDEN; kk += 32) {
        v16h a, b;
        if constexpr (sizeof(XT) == 4) {            // fp32 activations
            const float* ap = (const float*)X + (size_t)arow * HIDDEN + kk + hi * 8;
            float4 x0 = *(const float4*)(ap);
            float4 x1 = *(const float4*)(ap + 4);
            float4 x2 = *(const float4*)(ap + 16);
            float4 x3 = *(const float4*)(ap + 20);
            a[0]=(_Float16)x0.x;  a[1]=(_Float16)x0.y;  a[2]=(_Float16)x0.z;  a[3]=(_Float16)x0.w;
            a[4]=(_Float16)x1.x;  a[5]=(_Float16)x1.y;  a[6]=(_Float16)x1.z;  a[7]=(_Float16)x1.w;
            a[8]=(_Float16)x2.x;  a[9]=(_Float16)x2.y;  a[10]=(_Float16)x2.z; a[11]=(_Float16)x2.w;
            a[12]=(_Float16)x3.x; a[13]=(_Float16)x3.y; a[14]=(_Float16)x3.z; a[15]=(_Float16)x3.w;
        } else {                                    // f16 activations
            const _Float16* ap = (const _Float16*)X + (size_t)arow * HIDDEN + kk + hi * 8;
            v8h x0 = *(const v8h*)(ap);
            v8h x1 = *(const v8h*)(ap + 16);
#pragma unroll
            for (int h = 0; h < 8; ++h) { a[h] = x0[h]; a[h + 8] = x1[h]; }
        }
        {
            const float* bp = W + (size_t)bn * HIDDEN + kk + hi * 8;
            float4 w0 = *(const float4*)(bp);
            float4 w1 = *(const float4*)(bp + 4);
            float4 w2 = *(const float4*)(bp + 16);
            float4 w3 = *(const float4*)(bp + 20);
            b[0]=(_Float16)w0.x;  b[1]=(_Float16)w0.y;  b[2]=(_Float16)w0.z;  b[3]=(_Float16)w0.w;
            b[4]=(_Float16)w1.x;  b[5]=(_Float16)w1.y;  b[6]=(_Float16)w1.z;  b[7]=(_Float16)w1.w;
            b[8]=(_Float16)w2.x;  b[9]=(_Float16)w2.y;  b[10]=(_Float16)w2.z; b[11]=(_Float16)w2.w;
            b[12]=(_Float16)w3.x; b[13]=(_Float16)w3.y; b[14]=(_Float16)w3.z; b[15]=(_Float16)w3.w;
        }
        c = __builtin_amdgcn_wmma_f32_16x16x32_f16(false, a, false, b, (short)0, c, false, false);
    }
    const int col = nt * 16 + l15;
    const float bv = bias[col];
    if constexpr (TROUT) {
        v8h o;
#pragma unroll
        for (int r = 0; r < 8; ++r) o[r] = (_Float16)(c[r] + bv);
        *(v8h*)((_Float16*)Y + (size_t)col * M + mt * 16 + hi * 8) = o;   // 16B store
    } else {
#pragma unroll
        for (int r = 0; r < 8; ++r) {
            int row = mt * 16 + r + hi * 8;
            Y[(size_t)row * HIDDEN + col] = (YT)(c[r] + bv);
        }
    }
}

// ---- partial[ks][4096][256] = (T*inv) @ qh over K-chunk ks; single pass over T ----
// Streaming kernel at ~1 wave/SIMD: latency hidden *inside* the wave —
// A (T) loads rotated one k-step ahead; B tiles software-pipelined 8 deep.
__global__ void __launch_bounds__(128, 1)
tn_qh_gemm_kernel(const float* __restrict__ T, const float* __restrict__ inv,
                  const _Float16* __restrict__ qh_t, float* __restrict__ partial) {
    const int lane = threadIdx.x & 31;
    const int wid  = (blockIdx.x * blockDim.x + threadIdx.x) >> 5;   // 1024 waves
    const int mt = wid / KSPLIT;    // 0..255 : 16-row block
    const int ks = wid % KSPLIT;    // K chunk
    const int hi  = (lane & 16) ? 1 : 0;
    const int l15 = lane & 15;

    const int arow = mt * 16 + l15;
    const float ainv = inv[arow];
    const float* Trow = T + (size_t)arow * N_EDGES + (size_t)ks * KCHUNK;
    const _Float16* qcol = qh_t + (size_t)l15 * N_EDGES + (size_t)ks * KCHUNK + hi * 16;

    v8f acc[16];
#pragma unroll
    for (int i = 0; i < 16; ++i) acc[i] = (v8f){};

    // rotated A load: x* hold T data for the step about to be computed
    const float* ap = Trow + hi * 16;
    float4 x0 = *(const float4*)(ap);
    float4 x1 = *(const float4*)(ap + 4);
    float4 x2 = *(const float4*)(ap + 8);
    float4 x3 = *(const float4*)(ap + 12);

    for (int kk = 0; kk < KCHUNK; kk += 32) {
        __builtin_prefetch(Trow + kk + 1024, 0, 0);  // stream T (global_prefetch_b8)
        v16h a;
        a[0]=(_Float16)(x0.x*ainv);  a[1]=(_Float16)(x0.y*ainv);  a[2]=(_Float16)(x0.z*ainv);  a[3]=(_Float16)(x0.w*ainv);
        a[4]=(_Float16)(x1.x*ainv);  a[5]=(_Float16)(x1.y*ainv);  a[6]=(_Float16)(x1.z*ainv);  a[7]=(_Float16)(x1.w*ainv);
        a[8]=(_Float16)(x2.x*ainv);  a[9]=(_Float16)(x2.y*ainv);  a[10]=(_Float16)(x2.z*ainv); a[11]=(_Float16)(x2.w*ainv);
        a[12]=(_Float16)(x3.x*ainv); a[13]=(_Float16)(x3.y*ainv); a[14]=(_Float16)(x3.z*ainv); a[15]=(_Float16)(x3.w*ainv);

        // issue next step's A (T) loads now; 16 WMMAs below cover their latency
        if (kk + 32 < KCHUNK) {
            const float* apn = Trow + (kk + 32) + hi * 16;
            x0 = *(const float4*)(apn);
            x1 = *(const float4*)(apn + 4);
            x2 = *(const float4*)(apn + 8);
            x3 = *(const float4*)(apn + 12);
        }

        const _Float16* bbase = qcol + kk;           // tile nt at + nt*16*N_EDGES
        v16h bbuf[8];
#pragma unroll
        for (int j = 0; j < 8; ++j)
            bbuf[j] = *(const v16h*)(bbase + (size_t)j * 16 * N_EDGES);
#pragma unroll
        for (int j = 0; j < 8; ++j) {
            v16h bn = *(const v16h*)(bbase + (size_t)(j + 8) * 16 * N_EDGES);
            acc[j] = __builtin_amdgcn_wmma_f32_16x16x32_f16(false, a, false, bbuf[j], (short)0, acc[j], false, false);
            bbuf[j] = bn;
        }
#pragma unroll
        for (int j = 0; j < 8; ++j)
            acc[8 + j] = __builtin_amdgcn_wmma_f32_16x16x32_f16(false, a, false, bbuf[j], (short)0, acc[8 + j], false, false);
    }

    float* pout = partial + (size_t)ks * N_NODES * HIDDEN;
#pragma unroll
    for (int nt = 0; nt < 16; ++nt) {
        const int col = nt * 16 + l15;
#pragma unroll
        for (int r = 0; r < 8; ++r) {
            int row = mt * 16 + r + hi * 8;
            pout[(size_t)row * HIDDEN + col] = acc[nt][r];
        }
    }
}

// ---- qn = sum_ks partial * d^-0.5 -> f16 row-major ----
__global__ void qn_reduce_kernel(const float* __restrict__ partial, _Float16* __restrict__ qn) {
    const size_t i = (size_t)blockIdx.x * blockDim.x + threadIdx.x;   // over 1M/4 float4s
    const size_t n = (size_t)N_NODES * HIDDEN;
    const float scale = 0.17677669529663687f;
    float4 s0 = ((const float4*)partial)[i];
    float4 s1 = ((const float4*)(partial + n))[i];
    float4 s2 = ((const float4*)(partial + 2 * n))[i];
    float4 s3 = ((const float4*)(partial + 3 * n))[i];
    v4h o;
    o[0] = (_Float16)((s0.x + s1.x + s2.x + s3.x) * scale);
    o[1] = (_Float16)((s0.y + s1.y + s2.y + s3.y) * scale);
    o[2] = (_Float16)((s0.z + s1.z + s2.z + s3.z) * scale);
    o[3] = (_Float16)((s0.w + s1.w + s2.w + s3.w) * scale);
    ((v4h*)qn)[i] = o;
}

// ---------------- flash attention: per (head, 16-query block) wave ----------------
__global__ void __launch_bounds__(32 * ATT_WAVES, 1)
attention_kernel(const _Float16* __restrict__ qn, const _Float16* __restrict__ kh,
                 const _Float16* __restrict__ vh_t, _Float16* __restrict__ ctx) {
    const int lane  = threadIdx.x & 31;
    const int wslot = threadIdx.x >> 5;
    const int wid   = (blockIdx.x * blockDim.x + threadIdx.x) >> 5;
    const int head  = wid / (N_NODES / 16);
    const int nb    = wid % (N_NODES / 16);
    const int hoff  = head * D_HEAD;
    const int hi    = (lane & 16) ? 1 : 0;
    const int l15   = lane & 15;

    __shared__ _Float16 pls[ATT_WAVES][16 * 32];     // P: D-layout -> A-layout transpose
    _Float16* P = pls[wslot];

    v16h aq;
    {
        const _Float16* qp = qn + (size_t)(nb * 16 + l15) * HIDDEN + hoff + hi * 16;
        v8h q0 = *(const v8h*)(qp);
        v8h q1 = *(const v8h*)(qp + 8);
#pragma unroll
        for (int h = 0; h < 8; ++h) { aq[h] = q0[h]; aq[h + 8] = q1[h]; }
    }

    v8f c0 = {}, c1 = {};                            // ctx accum: 16 rows x (d 0..15 | 16..31)
    float rowmax[8], rowsum[8];
#pragma unroll
    for (int r = 0; r < 8; ++r) { rowmax[r] = -3.0e38f; rowsum[r] = 0.f; }

    for (int m0 = 0; m0 < N_NODES; m0 += 32) {
        v16h kb0 = *(const v16h*)(kh + (size_t)(m0 + l15) * HIDDEN + hoff + hi * 16);
        v16h kb1 = *(const v16h*)(kh + (size_t)(m0 + 16 + l15) * HIDDEN + hoff + hi * 16);
        v8f z = {};
        v8f s0 = __builtin_amdgcn_wmma_f32_16x16x32_f16(false, aq, false, kb0, (short)0, z, false, false);
        v8f s1 = __builtin_amdgcn_wmma_f32_16x16x32_f16(false, aq, false, kb1, (short)0, z, false, false);

#pragma unroll
        for (int r = 0; r < 8; ++r) {
            float mloc = redmax16(fmaxf(s0[r], s1[r]));
            float mnew = fmaxf(rowmax[r], mloc);
            float alpha = __expf(rowmax[r] - mnew);
            float p0 = __expf(s0[r] - mnew);
            float p1 = __expf(s1[r] - mnew);
            rowsum[r] = rowsum[r] * alpha + redsum16(p0 + p1);
            c0[r] *= alpha;
            c1[r] *= alpha;
            rowmax[r] = mnew;
            int prow = r + hi * 8;                   // D-layout row
            P[prow * 32 + l15]      = (_Float16)p0;
            P[prow * 32 + 16 + l15] = (_Float16)p1;
        }
        asm volatile("s_wait_dscnt 0" ::: "memory"); // P stores visible to A-layout reads

        v16h pf, vb0, vb1;
        {
            const _Float16* pp = P + l15 * 32 + hi * 16;
            v8h p0v = *(const v8h*)(pp);
            v8h p1v = *(const v8h*)(pp + 8);
#pragma unroll
            for (int h = 0; h < 8; ++h) { pf[h] = p0v[h]; pf[h + 8] = p1v[h]; }
        }
        vb0 = *(const v16h*)(vh_t + (size_t)(hoff + l15) * N_NODES + m0 + hi * 16);
        vb1 = *(const v16h*)(vh_t + (size_t)(hoff + 16 + l15) * N_NODES + m0 + hi * 16);

        c0 = __builtin_amdgcn_wmma_f32_16x16x32_f16(false, pf, false, vb0, (short)0, c0, false, false);
        c1 = __builtin_amdgcn_wmma_f32_16x16x32_f16(false, pf, false, vb1, (short)0, c1, false, false);
        asm volatile("s_wait_dscnt 0" ::: "memory"); // pf reads done before P is rewritten
    }

#pragma unroll
    for (int r = 0; r < 8; ++r) {
        int row = nb * 16 + r + hi * 8;
        float is = 1.f / rowsum[r];
        ctx[(size_t)row * HIDDEN + hoff + l15]      = (_Float16)(c0[r] * is);
        ctx[(size_t)row * HIDDEN + hoff + 16 + l15] = (_Float16)(c1[r] * is);
    }
}

extern "C" void kernel_launch(void* const* d_in, const int* in_sizes, int n_in,
                              void* d_out, int out_size, void* d_ws, size_t ws_size,
                              hipStream_t stream) {
    const float* q  = (const float*)d_in[0];
    const float* k  = (const float*)d_in[1];
    const float* v  = (const float*)d_in[2];
    const float* T  = (const float*)d_in[3];
    const float* Wq = (const float*)d_in[4];
    const float* bq = (const float*)d_in[5];
    const float* Wk = (const float*)d_in[6];
    const float* bk = (const float*)d_in[7];
    const float* Wv = (const float*)d_in[8];
    const float* bv = (const float*)d_in[9];
    const float* Wo = (const float*)d_in[10];
    const float* bo = (const float*)d_in[11];
    float* out = (float*)d_out;

    // workspace layout (~32.1 MB total)
    char* ws = (char*)d_ws;
    _Float16* qh_t = (_Float16*)ws; ws += (size_t)N_EDGES * HIDDEN * sizeof(_Float16);  // [256][16384]
    _Float16* kh   = (_Float16*)ws; ws += (size_t)N_NODES * HIDDEN * sizeof(_Float16);  // [4096][256]
    _Float16* vh_t = (_Float16*)ws; ws += (size_t)N_NODES * HIDDEN * sizeof(_Float16);  // [256][4096]
    _Float16* qnn  = (_Float16*)ws; ws += (size_t)N_NODES * HIDDEN * sizeof(_Float16);  // [4096][256]
    _Float16* ctx  = (_Float16*)ws; ws += (size_t)N_NODES * HIDDEN * sizeof(_Float16);  // [4096][256]
    float*    part = (float*)ws;    ws += (size_t)KSPLIT * N_NODES * HIDDEN * sizeof(float);
    float*    inv  = (float*)ws;

    // 1) 1/rowsum(T)
    rowsum_inv_kernel<<<N_NODES, 256, 0, stream>>>(T, inv);

    // 2) projections (fp32 in, f16 out); q and v written transposed
    {
        int waves = (N_EDGES / 16) * 16;   // 16384
        gemm256_wmma_kernel<float, _Float16, true><<<waves / 4, 128, 0, stream>>>(q, Wq, bq, qh_t, N_EDGES);
    }
    {
        int waves = (N_NODES / 16) * 16;   // 4096
        gemm256_wmma_kernel<float, _Float16, false><<<waves / 4, 128, 0, stream>>>(k, Wk, bk, kh, N_NODES);
        gemm256_wmma_kernel<float, _Float16, true><<<waves / 4, 128, 0, stream>>>(v, Wv, bv, vh_t, N_NODES);
    }

    // 3) single streaming pass over T: each wave covers all 256 output columns
    {
        int waves = (N_NODES / 16) * KSPLIT;   // 1024
        tn_qh_gemm_kernel<<<waves / 4, 128, 0, stream>>>(T, inv, qh_t, part);
        qn_reduce_kernel<<<(N_NODES * HIDDEN / 4) / 256, 256, 0, stream>>>(part, qnn);
    }

    // 4) flash attention per head
    {
        int waves = N_HEADS * (N_NODES / 16);  // 2048
        attention_kernel<<<waves / ATT_WAVES, 32 * ATT_WAVES, 0, stream>>>(qnn, kh, vh_t, ctx);
    }

    // 5) out = ctx @ Wo^T + bo  (f16 in, fp32 out)
    {
        int waves = (N_NODES / 16) * 16;       // 4096
        gemm256_wmma_kernel<_Float16, float, false><<<waves / 4, 128, 0, stream>>>(ctx, Wo, bo, out, N_NODES);
    }
}
